// BlurredNoise_35665408426088
// MI455X (gfx1250) — matrix operand order repaired and checked by reading, not compiled, not run on hardware
//
#include <hip/hip_runtime.h>

// ---------------------------------------------------------------------------
// BlurredNoise on MI455X (gfx1250):
// 16 independent GEMMs  (M=128 filters) x (N=4096 time) x (K=5000 taps)
// done as bf16-split (hi+lo) WMMA:  a*b ~= ah*bh + ah*bl + al*bh, fp32 accum.
// Compute-bound: 84 GFLOP vs ~40MB of traffic -> everything lives in L2.
// Noise planes are stored in 8 element-shift copies so every lane's Toeplitz
// B-fragment run is a pair of 16B-aligned b128 loads (20 VMEM / 12 WMMA per
// K=32 chunk).
// ---------------------------------------------------------------------------

typedef __attribute__((ext_vector_type(16))) __bf16        v16bf;
typedef __attribute__((ext_vector_type(8)))  float         v8f;
typedef __attribute__((ext_vector_type(8)))  unsigned int  v8u;

#define BC_ROWS  16          // batch*noise_channels
#define F_TOT    128         // filters
#define T_OUT    4096        // output time steps
#define KF       5000        // filter taps
#define KP       5024        // taps padded to multiple of 32 (157 chunks)
#define NCHUNK   157
#define IN_SEQ   9095        // input time steps
#define ND       4560        // dwords per noise-plane row (max access 4555)
#define NSHIFT   8           // element-shift copies of the noise planes

union BfCast { v8u u; v16bf b; };

static __device__ __forceinline__ v16bf mkfrag(uint4 a, uint4 b) {
  BfCast t; t.u = (v8u){a.x, a.y, a.z, a.w, b.x, b.y, b.z, b.w}; return t.b;
}
static __device__ __forceinline__ v8f wmma_bf16(v16bf a, v16bf b, v8f c) {
  return __builtin_amdgcn_wmma_f32_16x16x32_bf16(false, a, false, b, (short)0, c,
                                                 false, false);
}

// --- prep 1: filters -> bf16 hi/lo planes, zero-padded K to 5024 ------------
__global__ __launch_bounds__(256) void prep_filters(
    const float* __restrict__ blur,
    unsigned short* __restrict__ fh,
    unsigned short* __restrict__ fl) {
  int idx = blockIdx.x * 256 + threadIdx.x;
  if (idx >= F_TOT * KP) return;
  int f = idx / KP;
  int k = idx - f * KP;
  float x = (k < KF) ? blur[f * KF + k] : 0.0f;
  unsigned int bx = __float_as_uint(x);
  float hi = __uint_as_float(bx & 0xFFFF0000u);
  float r  = x - hi;                         // exact in fp32
  fh[idx] = (unsigned short)(bx >> 16);
  fl[idx] = (unsigned short)(__float_as_uint(r) >> 16);
}

// --- prep 2: noise -> bf16 hi/lo packed-pair planes, 8 shift copies ---------
// Plane s, row rw, dword i holds bf16 elements (2i+s, 2i+s+1) of the row, so
// a lane whose element offset is == s (mod 8) gets 16B-aligned b128 runs.
__global__ __launch_bounds__(256) void prep_noise(
    const float* __restrict__ noise,
    unsigned int* __restrict__ NH,
    unsigned int* __restrict__ NL) {
  int idx = blockIdx.x * 256 + threadIdx.x;
  if (idx >= BC_ROWS * ND) return;
  int rw = idx / ND;
  int i  = idx - rw * ND;
  unsigned int hb[NSHIFT + 1], lb[NSHIFT + 1];
#pragma unroll
  for (int j = 0; j <= NSHIFT; ++j) {
    int e = 2 * i + j;
    float x = (e < IN_SEQ) ? noise[rw * IN_SEQ + e] : 0.0f;
    unsigned int bx = __float_as_uint(x);
    float hi = __uint_as_float(bx & 0xFFFF0000u);
    float r  = x - hi;
    hb[j] = bx >> 16;
    lb[j] = __float_as_uint(r) >> 16;
  }
#pragma unroll
  for (int s = 0; s < NSHIFT; ++s) {
    NH[(s * BC_ROWS + rw) * ND + i] = hb[s] | (hb[s + 1] << 16);
    NL[(s * BC_ROWS + rw) * ND + i] = lb[s] | (lb[s + 1] << 16);
  }
}

// --- main: implicit-Toeplitz GEMM with bf16x3 WMMA --------------------------
// block = 256 threads = 8 waves; block -> (bc row, 64-wide time strip),
// wave w -> filter tile [16w, 16w+16); each wave: 4 N-tiles of 16.
__global__ __launch_bounds__(256) void conv_wmma(
    const unsigned short* __restrict__ fh,
    const unsigned short* __restrict__ fl,
    const unsigned int* __restrict__ NH,
    const unsigned int* __restrict__ NL,
    const float* __restrict__ oscale,
    float* __restrict__ out) {
  const int bc    = blockIdx.x >> 6;       // 0..15
  const int strip = blockIdx.x & 63;       // 0..63 (64 time points each)
  const int wave  = threadIdx.x >> 5;      // 0..7  == filter tile
  const int lane  = threadIdx.x & 31;
  const int hv    = lane >> 4;             // half-wave
  const int lm    = lane & 15;
  const int s     = lm & (NSHIFT - 1);     // shift-plane selector

  // A (filters, 16-bit A 16x32 layout): lane row = lm,
// K run [k0+16hv, +16) -> two aligned b128 per plane.
  const unsigned short* fa_h = fh + (wave * 16 + lm) * KP + 16 * hv;
  const unsigned short* fa_l = fl + (wave * 16 + lm) * KP + 16 * hv;

  // B (Toeplitz noise, 32x16 layout): lane col n = strip*64 + 16*nt + lm,
  // K run = 16 consecutive elements starting at e0 + 32c (+16 per nt).
  const int i0 = ((strip * 64 + lm + 16 * hv) - s) >> 1;   // == 0 (mod 4)
  const unsigned int* nh = NH + (s * BC_ROWS + bc) * ND + i0;
  const unsigned int* nl = NL + (s * BC_ROWS + bc) * ND + i0;

  v8f acc[4] = {v8f{}, v8f{}, v8f{}, v8f{}};

#pragma unroll 2
  for (int c = 0; c < NCHUNK; ++c) {
    const int k0 = c * 32;
    v16bf Ah = mkfrag(*(const uint4*)(fa_h + k0),
                      *(const uint4*)(fa_h + k0 + 8));
    v16bf Al = mkfrag(*(const uint4*)(fa_l + k0),
                      *(const uint4*)(fa_l + k0 + 8));
    const unsigned int* ph = nh + 16 * c;
    const unsigned int* pl = nl + 16 * c;
#pragma unroll
    for (int nt = 0; nt < 4; ++nt) {
      v16bf Bh = mkfrag(*(const uint4*)(ph + 8 * nt),
                        *(const uint4*)(ph + 8 * nt + 4));
      v16bf Bl = mkfrag(*(const uint4*)(pl + 8 * nt),
                        *(const uint4*)(pl + 8 * nt + 4));
      acc[nt] = wmma_bf16(Ah, Bh, acc[nt]);   // hi*hi
      acc[nt] = wmma_bf16(Ah, Bl, acc[nt]);   // hi*lo
      acc[nt] = wmma_bf16(Al, Bh, acc[nt]);   // lo*hi
    }
  }

  // D layout: VGPR r, lane -> row m = r + 8*hv, col n = lm.
  float sc[8];
#pragma unroll
  for (int r = 0; r < 8; ++r)
    sc[r] = oscale[wave * 16 + 8 * hv + r];  // feats * output_scale (NPF==1)

  const size_t obase =
      (size_t)(bc * F_TOT + wave * 16 + 8 * hv) * T_OUT + strip * 64 + lm;
#pragma unroll
  for (int nt = 0; nt < 4; ++nt)
#pragma unroll
    for (int r = 0; r < 8; ++r)
      out[obase + (size_t)r * T_OUT + nt * 16] = acc[nt][r] * sc[r];
}

// ---------------------------------------------------------------------------
extern "C" void kernel_launch(void* const* d_in, const int* in_sizes, int n_in,
                              void* d_out, int out_size, void* d_ws,
                              size_t ws_size, hipStream_t stream) {
  const float* noise  = (const float*)d_in[0];   // (2, 8, 9095) f32
  const float* blur   = (const float*)d_in[1];   // (128, 5000) f32
  const float* oscale = (const float*)d_in[2];   // (1, 128, 1) f32
  float* out = (float*)d_out;                    // (2, 1024, 4096) f32

  // workspace layout (~6.9 MB total):
  //   fh, fl : ushort[128*5024]                 filter hi/lo bf16 planes
  //   NH, NL : uint [8 shifts][16 rows][4560]   packed bf16-pair noise planes
  unsigned short* fh = (unsigned short*)d_ws;
  unsigned short* fl = fh + (size_t)F_TOT * KP;
  unsigned int*   NH = (unsigned int*)(fl + (size_t)F_TOT * KP);
  unsigned int*   NL = NH + (size_t)NSHIFT * BC_ROWS * ND;

  prep_filters<<<(F_TOT * KP + 255) / 256, 256, 0, stream>>>(blur, fh, fl);
  prep_noise<<<(BC_ROWS * ND + 255) / 256, 256, 0, stream>>>(noise, NH, NL);
  conv_wmma<<<dim3(BC_ROWS * 64), dim3(256), 0, stream>>>(fh, fl, NH, NL,
                                                          oscale, out);
}
